// MultiHeadSelfAttention_5480378270183
// MI455X (gfx1250) — matrix operand order repaired and checked by reading, not compiled, not run on hardware
//
#include <hip/hip_runtime.h>
#include <hip/hip_bf16.h>

typedef __attribute__((ext_vector_type(16))) _Float16 v16h;
typedef __attribute__((ext_vector_type(4)))  _Float16 v4h;
typedef __attribute__((ext_vector_type(8)))  float    v8f;

// Problem constants
#define BB   2
#define SS   2048
#define DD   1024
#define HH   16
#define HD   64
#define N3   3072

// ---------------------------------------------------------------------------
// WMMA helper (gfx1250: V_WMMA_F32_16X16X32_F16, wave32)
// ---------------------------------------------------------------------------
__device__ __forceinline__ v8f wmma_f16(v16h a, v16h b, v8f c) {
  return __builtin_amdgcn_wmma_f32_16x16x32_f16(false, a, false, b,
                                                (short)0, c, false, false);
}

// ---------------------------------------------------------------------------
// Fragment-linear LDS layout: one 16x32 (A) or 32x16 (B) f16 fragment = 512
// halves; lane L owns halves [L*16, L*16+16). A fragment load is one 32-byte
// vector load (2x ds_load_b128), no repacking.
//
// A-frag element (m 0..15, k 0..31): lane = m + (k&8 ? 16:0),
//                                    half = ((k&16)>>1) + (k&7)
// B-frag element (k 0..31, n 0..15): lane = n + (k&8 ? 16:0),
//                                    half = ((k&16)>>1) + (k&7)
// ---------------------------------------------------------------------------
__device__ __forceinline__ v16h frag_load(const _Float16* fragbase) {
  return *(const v16h*)(fragbase + (threadIdx.x & 31) * 16);
}

// Gather A-fragment from row-major global f16 (used once for Q)
__device__ __forceinline__ v16h load_a16_global(const _Float16* base, int ld) {
  const int lane = threadIdx.x & 31;
  const int row  = lane & 15;
  const int kg   = (lane >> 4) << 3;
  v16h a;
#pragma unroll
  for (int v = 0; v < 8; ++v) {
    const int kb = ((v & 4) << 2) + kg + ((v & 3) << 1);
    a[2 * v]     = base[row * ld + kb];
    a[2 * v + 1] = base[row * ld + kb + 1];
  }
  return a;
}

// ---------------------------------------------------------------------------
// Kernel 1: QKV projection. C[4096,3072] = x[4096,1024] @ Wqkv[1024,3072]+bqkv
// Scatters into Q/K/V f16 workspace, layout [B*H][S][hd]. Q pre-scaled 0.125.
// ---------------------------------------------------------------------------
__global__ void __launch_bounds__(256)
qkv_gemm_kernel(const float* __restrict__ x, const float* __restrict__ Wqkv,
                const float* __restrict__ bqkv, _Float16* __restrict__ qws,
                _Float16* __restrict__ kws, _Float16* __restrict__ vws) {
  __shared__ __align__(32) _Float16 Af[8 * 512];  // 8 m-subtiles
  __shared__ __align__(32) _Float16 Bf[8 * 512];  // 8 n-subtiles
  const int t    = threadIdx.x;
  const int w    = t >> 5;
  const int lane = t & 31;
  const int m0   = blockIdx.y * 128;
  const int n0   = blockIdx.x * 128;
  const int m_sub = (w & 3) * 2;   // 2 m-subtiles per wave
  const int n_sub = (w >> 2) * 4;  // 4 n-subtiles per wave

  v8f acc[2][4];
#pragma unroll
  for (int i = 0; i < 2; ++i)
#pragma unroll
    for (int j = 0; j < 4; ++j)
#pragma unroll
      for (int r = 0; r < 8; ++r) acc[i][j][r] = 0.0f;

  for (int kt = 0; kt < DD / 32; ++kt) {
    const int k0 = kt * 32;
    // A tile: 128x32, f32->f16, swizzled into fragment layout.
    // Thread loads 4 consecutive k for one row -> 4 contiguous halves.
#pragma unroll
    for (int p = 0; p < 4; ++p) {
      const int row = (t >> 3) + p * 32;
      const int col = (t & 7) * 4;
      float4 xv = *(const float4*)(x + (size_t)(m0 + row) * DD + k0 + col);
      const int lanei = (row & 15) + ((col & 8) ? 16 : 0);
      const int h0    = ((col & 16) >> 1) + (col & 7);
      v4h pk;
      pk[0] = (_Float16)xv.x; pk[1] = (_Float16)xv.y;
      pk[2] = (_Float16)xv.z; pk[3] = (_Float16)xv.w;
      *(v4h*)(&Af[(row >> 4) * 512 + lanei * 16 + h0]) = pk;
    }
    // B tile: 32x128. Thread loads a k-quad for one n (coalesced across t)
    // -> 4 contiguous halves in fragment layout.
#pragma unroll
    for (int p = 0; p < 4; ++p) {
      const int nr = t & 127;
      const int kb = ((t >> 7) + p * 2) * 4;
      const float* gb = Wqkv + (size_t)(k0 + kb) * N3 + n0 + nr;
      v4h pk;
      pk[0] = (_Float16)gb[0];
      pk[1] = (_Float16)gb[N3];
      pk[2] = (_Float16)gb[2 * N3];
      pk[3] = (_Float16)gb[3 * N3];
      const int lanei = (nr & 15) + ((kb & 8) ? 16 : 0);
      *(v4h*)(&Bf[(nr >> 4) * 512 + lanei * 16 + ((kb & 16) >> 1) + (kb & 7)]) = pk;
    }
    __syncthreads();
    if (kt + 1 < DD / 32) {   // global_prefetch_b8 of next tiles
      __builtin_prefetch(x + (size_t)(m0 + (t >> 3)) * DD + k0 + 32, 0, 1);
      __builtin_prefetch(Wqkv + (size_t)(k0 + 32 + (t >> 5)) * N3 + n0, 0, 1);
    }
    v16h af0 = frag_load(&Af[(m_sub + 0) * 512]);
    v16h af1 = frag_load(&Af[(m_sub + 1) * 512]);
#pragma unroll
    for (int j = 0; j < 4; ++j) {
      v16h bf = frag_load(&Bf[(n_sub + j) * 512]);
      acc[0][j] = wmma_f16(af0, bf, acc[0][j]);
      acc[1][j] = wmma_f16(af1, bf, acc[1][j]);
    }
    __syncthreads();
  }

  const int colL  = lane & 15;
  const int rhalf = (lane >> 4) * 8;
#pragma unroll
  for (int i = 0; i < 2; ++i)
#pragma unroll
    for (int j = 0; j < 4; ++j)
#pragma unroll
      for (int r = 0; r < 8; ++r) {
        const int rowg = m0 + (m_sub + i) * 16 + r + rhalf;
        const int colg = n0 + (n_sub + j) * 16 + colL;
        float val = acc[i][j][r] + bqkv[colg];
        const int which = colg >> 10;       // 0=q 1=k 2=v
        const int d  = colg & 1023;
        const int h  = d >> 6, dd = d & 63;
        const int b  = rowg >> 11, s = rowg & 2047;
        const size_t dst = ((size_t)(b * HH + h) * SS + s) * HD + dd;
        if (which == 0)      qws[dst] = (_Float16)(val * 0.125f);
        else if (which == 1) kws[dst] = (_Float16)val;
        else                 vws[dst] = (_Float16)val;
      }
}

// ---------------------------------------------------------------------------
// Kernel 2: flash attention. grid = (S/128, B*H). Each wave owns 16 q rows.
// K tile is streamed global->LDS with GLOBAL_LOAD_ASYNC_TO_LDS_B128 directly
// into B-fragment layout (ASYNCcnt / s_wait_asynccnt).
// ---------------------------------------------------------------------------
__global__ void __launch_bounds__(256)
attn_kernel(const _Float16* __restrict__ qws, const _Float16* __restrict__ kws,
            const _Float16* __restrict__ vws, const float* __restrict__ mask,
            const float* __restrict__ bias, _Float16* __restrict__ aws) {
  __shared__ __align__(32) _Float16 Kf[4 * 512];  // frag f = dh*2 + c
  __shared__ __align__(32) _Float16 Vf[4 * 512];  // frag j (d-group)
  __shared__ __align__(32) _Float16 Pf[8 * 512];  // per-wave P fragment
  const int t = threadIdx.x, w = t >> 5, lane = t & 31;
  const int bh    = blockIdx.y;
  const int batch = bh >> 4;
  const int head  = bh & 15;
  const int q0    = blockIdx.x * 128 + w * 16;

  const _Float16* qbase = qws + ((size_t)bh * SS + q0) * HD;
  v16h aq0 = load_a16_global(qbase, HD);
  v16h aq1 = load_a16_global(qbase + 32, HD);

  float mrow[8], lrow[8];
  v8f o[4];
#pragma unroll
  for (int r = 0; r < 8; ++r) { mrow[r] = -1e30f; lrow[r] = 0.0f; }
#pragma unroll
  for (int j = 0; j < 4; ++j)
#pragma unroll
    for (int r = 0; r < 8; ++r) o[j][r] = 0.0f;

  const int colL  = lane & 15;
  const int rhalf = (lane >> 4) * 8;
  const int kk    = t >> 3;          // k-row handled by this thread
  const int d0    = (t & 7) * 8;     // 8-d slice handled by this thread

  // Per-thread LDS destination for the async K copy: 16 contiguous bytes in
  // B-fragment layout for QK^T (K-dim = d, col = k-row).
  const int kfrag  = ((d0 >> 5) << 1) + (kk >> 4);
  const int klane  = (kk & 15) + ((d0 & 8) ? 16 : 0);
  const unsigned k_lds =
      (unsigned)(size_t)&Kf[kfrag * 512 + klane * 16 + ((d0 & 16) >> 1)];

  for (int kt = 0; kt < SS / 32; ++kt) {
    const int k0 = kt * 32;
    // --- K tile: async global->LDS, one b128 per thread ---
    {
      const _Float16* gK = kws + ((size_t)bh * SS + k0 + kk) * HD + d0;
      asm volatile("global_load_async_to_lds_b128 %0, %1, off"
                   :: "v"(k_lds), "v"(gK) : "memory");
    }
    // --- V tile: b128 global load + fragment-layout scatter ---
    {
      float4 vv = *(const float4*)(vws + ((size_t)bh * SS + k0 + kk) * HD + d0);
      const _Float16* vh = (const _Float16*)&vv;
#pragma unroll
      for (int i = 0; i < 8; ++i) {
        const int d = d0 + i;
        const int lv = (d & 15) + ((kk & 8) ? 16 : 0);
        Vf[(d >> 4) * 512 + lv * 16 + ((kk & 16) >> 1) + (kk & 7)] = vh[i];
      }
    }
    asm volatile("s_wait_asynccnt 0x0" ::: "memory");
    __syncthreads();

    // scores = Q @ K^T (pre-scaled) + mask + bias  -> two 16x16 C-frags
    v8f sc[2];
#pragma unroll
    for (int c = 0; c < 2; ++c) {
      v8f s;
#pragma unroll
      for (int r = 0; r < 8; ++r) s[r] = 0.0f;
      s = wmma_f16(aq0, frag_load(&Kf[(0 + c) * 512]), s);
      s = wmma_f16(aq1, frag_load(&Kf[(2 + c) * 512]), s);
#pragma unroll
      for (int r = 0; r < 8; ++r) {
        const int qg = q0 + r + rhalf;
        const int kg = k0 + c * 16 + colL;
        s[r] += mask[(size_t)qg * SS + kg] +
                bias[((size_t)batch * SS + qg) * SS + kg];
      }
      sc[c] = s;
    }

    // online softmax; row stats replicated across each 16-lane half
    float corr[8];
#pragma unroll
    for (int r = 0; r < 8; ++r) {
      float mx = fmaxf(sc[0][r], sc[1][r]);
#pragma unroll
      for (int off = 1; off < 16; off <<= 1)
        mx = fmaxf(mx, __shfl_xor(mx, off, 32));
      const float mnew = fmaxf(mrow[r], mx);
      corr[r] = __expf(mrow[r] - mnew);
      mrow[r] = mnew;
    }
#pragma unroll
    for (int c = 0; c < 2; ++c)
#pragma unroll
      for (int r = 0; r < 8; ++r) sc[c][r] = __expf(sc[c][r] - mrow[r]);
#pragma unroll
    for (int r = 0; r < 8; ++r) {
      float sm = sc[0][r] + sc[1][r];
#pragma unroll
      for (int off = 1; off < 16; off <<= 1) sm += __shfl_xor(sm, off, 32);
      lrow[r] = lrow[r] * corr[r] + sm;
    }

    // P (C-layout) -> wave-private LDS in A-fragment layout
    _Float16* Pw = &Pf[w * 512];
#pragma unroll
    for (int c = 0; c < 2; ++c)
#pragma unroll
      for (int r = 0; r < 8; ++r) {
        const int M = r + rhalf;
        const int k = c * 16 + colL;
        const int ldst = M + ((k & 8) ? 16 : 0);
        const int hh   = ((k & 16) >> 1) + (k & 7);
        Pw[ldst * 16 + hh] = (_Float16)sc[c][r];
      }
#pragma unroll
    for (int j = 0; j < 4; ++j)
#pragma unroll
      for (int r = 0; r < 8; ++r) o[j][r] *= corr[r];
    v16h ap = frag_load(Pw);
#pragma unroll
    for (int j = 0; j < 4; ++j)
      o[j] = wmma_f16(ap, frag_load(&Vf[j * 512]), o[j]);
    __syncthreads();
  }

  // normalize and write attention output as f16 [B*S, D]
#pragma unroll
  for (int j = 0; j < 4; ++j)
#pragma unroll
    for (int r = 0; r < 8; ++r) {
      const int qg = q0 + r + rhalf;
      const float val = o[j][r] / lrow[r];
      aws[((size_t)batch * SS + qg) * DD + head * HD + j * 16 + colL] =
          (_Float16)val;
    }
}

// ---------------------------------------------------------------------------
// Kernel 3: out = attn[4096,1024] @ Wout[1024,1024] + bout  (f32 output)
// ---------------------------------------------------------------------------
__global__ void __launch_bounds__(256)
out_gemm_kernel(const _Float16* __restrict__ aws, const float* __restrict__ Wout,
                const float* __restrict__ bout, float* __restrict__ out) {
  __shared__ __align__(32) _Float16 Af[8 * 512];
  __shared__ __align__(32) _Float16 Bf[8 * 512];
  const int t = threadIdx.x, w = t >> 5, lane = t & 31;
  const int m0 = blockIdx.y * 128, n0 = blockIdx.x * 128;
  const int m_sub = (w & 3) * 2, n_sub = (w >> 2) * 4;

  v8f acc[2][4];
#pragma unroll
  for (int i = 0; i < 2; ++i)
#pragma unroll
    for (int j = 0; j < 4; ++j)
#pragma unroll
      for (int r = 0; r < 8; ++r) acc[i][j][r] = 0.0f;

  for (int kt = 0; kt < DD / 32; ++kt) {
    const int k0 = kt * 32;
    // A tile already f16: thread copies 8 contiguous halves -> contiguous
    // 16 bytes in fragment layout.
#pragma unroll
    for (int p = 0; p < 2; ++p) {
      const int row = (t >> 2) + p * 64;
      const int col = (t & 3) * 8;
      float4 av = *(const float4*)(aws + (size_t)(m0 + row) * DD + k0 + col);
      const int lanei = (row & 15) + ((col & 8) ? 16 : 0);
      const int h0    = (col & 16) >> 1;
      *(float4*)(&Af[(row >> 4) * 512 + lanei * 16 + h0]) = av;
    }
    // B tile: k-quad per n, f32->f16
#pragma unroll
    for (int p = 0; p < 4; ++p) {
      const int nr = t & 127;
      const int kb = ((t >> 7) + p * 2) * 4;
      const float* gb = Wout + (size_t)(k0 + kb) * DD + n0 + nr;
      v4h pk;
      pk[0] = (_Float16)gb[0];
      pk[1] = (_Float16)gb[DD];
      pk[2] = (_Float16)gb[2 * DD];
      pk[3] = (_Float16)gb[3 * DD];
      const int lanei = (nr & 15) + ((kb & 8) ? 16 : 0);
      *(v4h*)(&Bf[(nr >> 4) * 512 + lanei * 16 + ((kb & 16) >> 1) + (kb & 7)]) = pk;
    }
    __syncthreads();
    if (kt + 1 < DD / 32) {
      __builtin_prefetch(aws + (size_t)(m0 + (t >> 2)) * DD + k0 + 32, 0, 1);
      __builtin_prefetch(Wout + (size_t)(k0 + 32 + (t >> 5)) * DD + n0, 0, 1);
    }
    v16h af0 = frag_load(&Af[(m_sub + 0) * 512]);
    v16h af1 = frag_load(&Af[(m_sub + 1) * 512]);
#pragma unroll
    for (int j = 0; j < 4; ++j) {
      v16h bf = frag_load(&Bf[(n_sub + j) * 512]);
      acc[0][j] = wmma_f16(af0, bf, acc[0][j]);
      acc[1][j] = wmma_f16(af1, bf, acc[1][j]);
    }
    __syncthreads();
  }

  const int colL  = lane & 15;
  const int rhalf = (lane >> 4) * 8;
#pragma unroll
  for (int i = 0; i < 2; ++i)
#pragma unroll
    for (int j = 0; j < 4; ++j)
#pragma unroll
      for (int r = 0; r < 8; ++r) {
        const int rowg = m0 + (m_sub + i) * 16 + r + rhalf;
        const int colg = n0 + (n_sub + j) * 16 + colL;
        out[(size_t)rowg * DD + colg] = acc[i][j][r] + bout[colg];
      }
}

// ---------------------------------------------------------------------------
extern "C" void kernel_launch(void* const* d_in, const int* in_sizes, int n_in,
                              void* d_out, int out_size, void* d_ws,
                              size_t ws_size, hipStream_t stream) {
  const float* x    = (const float*)d_in[0];
  const float* mask = (const float*)d_in[1];
  const float* bias = (const float*)d_in[2];
  const float* Wqkv = (const float*)d_in[3];
  const float* bqkv = (const float*)d_in[4];
  const float* Wout = (const float*)d_in[5];
  const float* bout = (const float*)d_in[6];
  float* out = (float*)d_out;

  const size_t QKV_ELEMS = (size_t)BB * HH * SS * HD;  // 4,194,304
  _Float16* qws = (_Float16*)d_ws;
  _Float16* kws = qws + QKV_ELEMS;
  _Float16* vws = kws + QKV_ELEMS;
  _Float16* aws = vws + QKV_ELEMS;

  qkv_gemm_kernel<<<dim3(N3 / 128, (BB * SS) / 128), 256, 0, stream>>>(
      x, Wqkv, bqkv, qws, kws, vws);
  attn_kernel<<<dim3(SS / 128, BB * HH), 256, 0, stream>>>(
      qws, kws, vws, mask, bias, aws);
  out_gemm_kernel<<<dim3(DD / 128, (BB * SS) / 128), 256, 0, stream>>>(
      aws, Wout, bout, out);
}
